// GSA_LSA_Block_12481174962664
// MI455X (gfx1250) — compile-verified, matrix operand order checked by reading, not run on hardware
//
#include <hip/hip_runtime.h>

// ---------------------------------------------------------------------------
// CDNA5 (gfx1250) dual-attention block: bf16 WMMA (v_wmma_f32_16x16x32_bf16),
// f32 softmax, LDS-staged operand strips, wave32.
// ---------------------------------------------------------------------------

typedef __attribute__((ext_vector_type(16))) __bf16 v16bf;
typedef __attribute__((ext_vector_type(8)))  __bf16 v8bf;
typedef __attribute__((ext_vector_type(8)))  float  v8f;

#define BB    4
#define NTOK  1024
#define CTOT  960

// CDNA5 16-bit A-fragment K index for element e on a lane:
// lanes 0-15 hold K {0..7,16..23}, lanes 16-31 hold K {8..15,24..31}.
// B-fragment (32x16, KxN) assumed symmetric: lane = N column, same K pattern.
__device__ __forceinline__ int kpat(int e, int lane) {
  return e + ((e >> 3) << 3) + ((lane >> 4) << 3);
}

// A-fragment from an LDS bf16 row: two contiguous 16B chunks -> 2x ds_load_b128.
__device__ __forceinline__ v16bf fragA_lds(const __bf16* row, int base, int lane) {
  int off = base + ((lane >> 4) << 3);
  v8bf lo = *(const v8bf*)(row + off);
  v8bf hi = *(const v8bf*)(row + off + 16);
  v16bf a;
#pragma unroll
  for (int e = 0; e < 8; ++e) { a[e] = lo[e]; a[8 + e] = hi[e]; }
  return a;
}

// A-fragment from an LDS f32 row: 4x float4 LDS loads + bf16 convert.
__device__ __forceinline__ v16bf fragA_ldsf(const float* row, int base, int lane) {
  int off = base + ((lane >> 4) << 3);
  const float4 a0 = *(const float4*)(row + off);
  const float4 a1 = *(const float4*)(row + off + 4);
  const float4 a2 = *(const float4*)(row + off + 16);
  const float4 a3 = *(const float4*)(row + off + 20);
  v16bf a;
  a[0]  = (__bf16)a0.x; a[1]  = (__bf16)a0.y; a[2]  = (__bf16)a0.z; a[3]  = (__bf16)a0.w;
  a[4]  = (__bf16)a1.x; a[5]  = (__bf16)a1.y; a[6]  = (__bf16)a1.z; a[7]  = (__bf16)a1.w;
  a[8]  = (__bf16)a2.x; a[9]  = (__bf16)a2.y; a[10] = (__bf16)a2.z; a[11] = (__bf16)a2.w;
  a[12] = (__bf16)a3.x; a[13] = (__bf16)a3.y; a[14] = (__bf16)a3.z; a[15] = (__bf16)a3.w;
  return a;
}

// B-fragment where the contraction runs along contiguous channels:
// element e -> row[GS[e/4] + lb + e%4], 16B-aligned float4 loads.
__device__ __forceinline__ v16bf fragB_full(const float* row, const float* w,
                                            const float* bia, int lb) {
  v16bf r;
  const int GS[4] = {0, 4, 16, 20};
#pragma unroll
  for (int eg = 0; eg < 4; ++eg) {
    const float4 x4 = *(const float4*)(row + GS[eg] + lb);
    const float4 w4 = *(const float4*)(w   + GS[eg] + lb);
    const float4 b4 = *(const float4*)(bia + GS[eg] + lb);
    r[eg * 4 + 0] = (__bf16)(x4.x * w4.x + b4.x);
    r[eg * 4 + 1] = (__bf16)(x4.y * w4.y + b4.y);
    r[eg * 4 + 2] = (__bf16)(x4.z * w4.z + b4.z);
    r[eg * 4 + 3] = (__bf16)(x4.w * w4.w + b4.w);
  }
  return r;
}

// Tail (k0 == 224 of a 240-channel contraction): e-groups 0,1 in range,
// groups 2,3 (kk >= 240) are compile-time zero (pair with zero-padded A).
__device__ __forceinline__ v16bf fragB_tail(const float* row, const float* w,
                                            const float* bia, int lb) {
  v16bf r;
  const int GS[2] = {0, 4};
#pragma unroll
  for (int eg = 0; eg < 2; ++eg) {
    const float4 x4 = *(const float4*)(row + GS[eg] + lb);
    const float4 w4 = *(const float4*)(w   + GS[eg] + lb);
    const float4 b4 = *(const float4*)(bia + GS[eg] + lb);
    r[eg * 4 + 0] = (__bf16)(x4.x * w4.x + b4.x);
    r[eg * 4 + 1] = (__bf16)(x4.y * w4.y + b4.y);
    r[eg * 4 + 2] = (__bf16)(x4.z * w4.z + b4.z);
    r[eg * 4 + 3] = (__bf16)(x4.w * w4.w + b4.w);
  }
#pragma unroll
  for (int e = 8; e < 16; ++e) r[e] = (__bf16)0.f;
  return r;
}

struct DWArgs {
  const float* wq[2][4]; const float* bq[2][4];
  const float* wk[2][4]; const float* bk[2][4];
  const float* wv[2][4]; const float* bv[2][4];
};

// ---------------------------------------------------------------------------
// LayerNorm over f channels; writes into concatenated [B,N,960] buffer.
// ---------------------------------------------------------------------------
__global__ void ln_kernel(const float* __restrict__ src, int srcStride,
                          const float* __restrict__ gamma,
                          const float* __restrict__ beta,
                          float* __restrict__ dstcat, int coff, int f) {
  int r = blockIdx.x * blockDim.x + threadIdx.x;
  if (r >= BB * NTOK) return;
  const float* x = src + (size_t)r * srcStride;
  float s = 0.f, ss = 0.f;
  for (int c = 0; c < f; ++c) { float v = x[c]; s += v; ss += v * v; }
  float mean = s / (float)f;
  float var  = ss / (float)f - mean * mean;
  float inv  = rsqrtf(fmaxf(var, 0.f) + 1e-6f);
  float* d = dstcat + (size_t)r * CTOT + coff;
  for (int c = 0; c < f; ++c)
    d[c] = (x[c] - mean) * inv * gamma[c] + beta[c];
}

// ---------------------------------------------------------------------------
// Channel attention: one wave per (g, 16-row q-tile, b, h, variant).
//   S[16,240] = scale * sum_n Q K   (K-contraction = 1024 tokens)
//   softmax over 240, O[16,1024] = P @ V (contraction 240 padded to 256)
// The scramble permutation is affine in the contraction variable, so its
// div/mod is hoisted: one division per fragment element (phase 1) or per
// j-tile (phase 3), then constant strides / single conditional wrap.
// ---------------------------------------------------------------------------
__global__ __launch_bounds__(32)
void chan_attn_kernel(DWArgs args, const float* __restrict__ xcin,
                      float* __restrict__ og, float* __restrict__ ol) {
  int bx = blockIdx.x;                 // 0..14 -> (group, mtile)
  int g, mt;
  if (bx < 1)      { g = 0; mt = bx; }
  else if (bx < 3) { g = 1; mt = bx - 1; }
  else if (bx < 7) { g = 2; mt = bx - 3; }
  else             { g = 3; mt = bx - 7; }
  int by = blockIdx.y;                 // 0..31
  int b = by & 3, h = (by >> 2) & 3, var = by >> 4;
  int f = 64 << g, dq = 16 << g, coff = 64 * ((1 << g) - 1);
  int lane = threadIdx.x;
  bool scram = (var == 1);

  const float* pwq = args.wq[var][g]; const float* pbq = args.bq[var][g];
  const float* pwk = args.wk[var][g]; const float* pbk = args.bk[var][g];
  const float* pwv = args.wv[var][g]; const float* pbv = args.bv[var][g];
  float* out = var ? ol : og;

  __shared__ __bf16 Qs[16][NTOK];      // dw(Q) strip (scramble folded in)
  __shared__ float  S[16][256];
  __shared__ float  rstat[16];

  const float scale_c = rsqrtf((float)CTOT);
  const int   m  = lane & 15;
  const int   lb = (lane >> 4) << 3;

  // ---- phase 0: stage dw(Q)[m=0..15, n=0..1023] into LDS ----
  if (!scram) {
    // 16 q-channels are contiguous in memory: 1 token per lane, 4x float4.
    int qc0 = h * dq + mt * 16;
    const float4* wp = (const float4*)(pwq + qc0);
    const float4* bp = (const float4*)(pbq + qc0);
    float4 w4[4], b4[4];
#pragma unroll
    for (int q = 0; q < 4; ++q) { w4[q] = wp[q]; b4[q] = bp[q]; }
    for (int n = lane; n < NTOK; n += 32) {
      const float4* xp =
          (const float4*)(xcin + (size_t)(b * NTOK + n) * CTOT + coff + qc0);
#pragma unroll
      for (int q = 0; q < 4; ++q) {
        float4 x4 = xp[q];
        Qs[q * 4 + 0][n] = (__bf16)(x4.x * w4[q].x + b4[q].x);
        Qs[q * 4 + 1][n] = (__bf16)(x4.y * w4[q].y + b4[q].y);
        Qs[q * 4 + 2][n] = (__bf16)(x4.z * w4[q].z + b4[q].z);
        Qs[q * 4 + 3][n] = (__bf16)(x4.w * w4[q].w + b4[q].w);
      }
    }
  } else {
    // scramble trick: n += 4 steps idx by 4f -> quotient +1, rem const.
    for (int pi = lane; pi < 64; pi += 32) {
      int mm = pi & 15, p = pi >> 4;   // p = n phase 0..3
      int qcl = h * dq + mt * 16 + mm;
      int idx = (b * NTOK + p) * f + qcl;
      int cb  = 4 * f;
      int n0  = idx / cb; int rem = idx - n0 * cb;
      int c0  = rem >> 2, b0i = rem & 3;
      const float* ptr = xcin + (size_t)(b0i * NTOK + n0) * CTOT + coff + c0;
      float wE = pwq[c0], bE = pbq[c0];
      for (int i = 0; i < 256; ++i) {
        Qs[mm][p + 4 * i] = (__bf16)(*ptr * wE + bE);
        ptr += CTOT;
      }
    }
  }
  __syncthreads();

  // ---- phase 1: S[m][dk] (dk = 240 key channels, 15 tiles of 16) ----
  for (int t = 0; t < 15; ++t) {
    int kch = h * 240 + t * 16 + m;    // absolute key channel for B lane
    // per-element pointer/weight precompute (division hoisted out of n-loop;
    // n += 32 steps idx by 8*(4*960) -> quotient +8, rem const)
    const float* bp[16]; float bw[16], bc[16];
#pragma unroll
    for (int e = 0; e < 16; ++e) {
      int n = kpat(e, lane);
      if (!scram) {
        bp[e] = xcin + (size_t)(b * NTOK + n) * CTOT + kch;
        bw[e] = pwk[kch]; bc[e] = pbk[kch];
      } else {
        int idx = (b * NTOK + n) * CTOT + kch;
        int n0  = idx / (4 * CTOT); int rem = idx - n0 * (4 * CTOT);
        int c0  = rem >> 2, b0i = rem & 3;
        bp[e] = xcin + (size_t)(b0i * NTOK + n0) * CTOT + c0;
        bw[e] = pwk[c0]; bc[e] = pbk[c0];
      }
    }
    const int bstr = scram ? 8 * CTOT : 32 * CTOT;
    v8f acc = {};
    for (int n0 = 0; n0 < NTOK; n0 += 32) {
      v16bf a = fragA_lds(&Qs[m][0], n0, lane);
      v16bf bm;
#pragma unroll
      for (int e = 0; e < 16; ++e) {
        bm[e] = (__bf16)(*bp[e] * bw[e] + bc[e]);
        bp[e] += bstr;
      }
      acc = __builtin_amdgcn_wmma_f32_16x16x32_bf16(
          false, a, false, bm, (short)0, acc, false, false);
    }
    int ncol = t * 16 + m;
#pragma unroll
    for (int r = 0; r < 8; ++r) S[r + lb][ncol] = acc[r] * scale_c;
  }
  // zero-pad P columns 240..255
  for (int i = lane; i < 16 * 16; i += 32) S[i >> 4][240 + (i & 15)] = 0.f;
  __syncthreads();

  // ---- phase 2: softmax over 240 (1/sum folded into epilogue) ----
  if (lane < 16) {
    float mx = -1e30f;
    for (int c = 0; c < 240; ++c) mx = fmaxf(mx, S[lane][c]);
    float sum = 0.f;
    for (int c = 0; c < 240; ++c) {
      float e = __expf(S[lane][c] - mx);
      S[lane][c] = e;
      sum += e;
    }
    rstat[lane] = 1.f / sum;
  }
  __syncthreads();

  // ---- phase 3: O[m][tok] = (P @ V)[m][tok] / rowsum ----
  for (int j = 0; j < 64; ++j) {
    int tok = j * 16 + m;              // token column for B lane
    v8f acc = {};
    if (!scram) {
      const float* row = xcin + (size_t)(b * NTOK + tok) * CTOT + h * 240;
      const float* wr  = pwv + h * 240;
      const float* br  = pbv + h * 240;
      for (int k0 = 0; k0 < 224; k0 += 32) {
        v16bf a  = fragA_ldsf(&S[m][0], k0, lane);
        v16bf bm = fragB_full(row + k0, wr + k0, br + k0, lb);
        acc = __builtin_amdgcn_wmma_f32_16x16x32_bf16(
            false, a, false, bm, (short)0, acc, false, false);
      }
      v16bf a  = fragA_ldsf(&S[m][0], 224, lane);
      v16bf bm = fragB_tail(row + 224, wr + 224, br + 224, lb);
      acc = __builtin_amdgcn_wmma_f32_16x16x32_bf16(
          false, a, false, bm, (short)0, acc, false, false);
    } else {
      // one division per j; per element: single conditional wrap
      int idx0 = (b * NTOK + tok) * CTOT + h * 240;
      int n0b  = idx0 / (4 * CTOT);
      int remb = idx0 - n0b * (4 * CTOT);
      for (int k0 = 0; k0 < 224; k0 += 32) {
        v16bf a = fragA_ldsf(&S[m][0], k0, lane);
        v16bf bm;
#pragma unroll
        for (int e = 0; e < 16; ++e) {
          int d   = k0 + kpat(e, lane);
          int rem = remb + d, n0v = n0b;
          if (rem >= 4 * CTOT) { rem -= 4 * CTOT; ++n0v; }
          int c0 = rem >> 2, b0i = rem & 3;
          bm[e] = (__bf16)(xcin[(size_t)(b0i * NTOK + n0v) * CTOT + c0]
                           * pwv[c0] + pbv[c0]);
        }
        acc = __builtin_amdgcn_wmma_f32_16x16x32_bf16(
            false, a, false, bm, (short)0, acc, false, false);
      }
      // tail k0=224: e<8 -> kk = 224..239 in range; e>=8 -> kk>=240 zero
      v16bf a = fragA_ldsf(&S[m][0], 224, lane);
      v16bf bm;
#pragma unroll
      for (int e = 0; e < 8; ++e) {
        int d   = 224 + kpat(e, lane);
        int rem = remb + d, n0v = n0b;
        if (rem >= 4 * CTOT) { rem -= 4 * CTOT; ++n0v; }
        int c0 = rem >> 2, b0i = rem & 3;
        bm[e] = (__bf16)(xcin[(size_t)(b0i * NTOK + n0v) * CTOT + c0]
                         * pwv[c0] + pbv[c0]);
      }
#pragma unroll
      for (int e = 8; e < 16; ++e) bm[e] = (__bf16)0.f;
      acc = __builtin_amdgcn_wmma_f32_16x16x32_bf16(
          false, a, false, bm, (short)0, acc, false, false);
    }
    int col = j * 16 + m;
#pragma unroll
    for (int r = 0; r < 8; ++r) {
      int mm = r + lb;
      out[(size_t)(b * NTOK + col) * CTOT + coff + h * dq + mt * 16 + mm] =
          acc[r] * rstat[mm];
    }
  }
}

// ---------------------------------------------------------------------------
// xs2 = x + (og + ol) * wproj + bproj   (channel block residual)
// ---------------------------------------------------------------------------
__global__ void combine_chan_kernel(const float* __restrict__ xg,
                                    const float* __restrict__ og,
                                    const float* __restrict__ ol,
                                    const float* __restrict__ w,
                                    const float* __restrict__ bia,
                                    float* __restrict__ xs2, int coff, int f) {
  int i = blockIdx.x * blockDim.x + threadIdx.x;
  int tot = BB * NTOK * f;
  if (i >= tot) return;
  int r = i / f, c = i - r * f;
  size_t k = (size_t)r * CTOT + coff + c;
  xs2[k] = xg[i] + (og[k] + ol[k]) * w[c] + bia[c];
}

// ---------------------------------------------------------------------------
// Spatial attention: 4 waves per (qtile of 16 rows, g, b, h, variant).
// Full [16 x 1024] score strip in LDS (bf16), two-pass softmax.
// ---------------------------------------------------------------------------
__global__ __launch_bounds__(128)
void spat_attn_kernel(DWArgs args, const float* __restrict__ xsin,
                      float* __restrict__ og, float* __restrict__ ol) {
  int qt = blockIdx.x;                    // 0..63 query tiles
  int y  = blockIdx.y;                    // 0..127
  int g = y & 3, b = (y >> 2) & 3, h = (y >> 4) & 3, var = (y >> 6) & 1;
  int dv = 16 << g, coff = 64 * ((1 << g) - 1);
  int tid = threadIdx.x, lane = tid & 31, wid = tid >> 5;

  const float* pwq = args.wq[var][g]; const float* pbq = args.bq[var][g];
  const float* pwk = args.wk[var][g]; const float* pbk = args.bk[var][g];
  const float* pwv = args.wv[var][g]; const float* pbv = args.bv[var][g];
  float* out = var ? ol : og;
  const float scale = rsqrtf(240.f);      // (C_TOT/heads)^-0.5

  __shared__ __bf16 Qs[16][256];          // dw(Q) tile, padded 240->256
  __shared__ __bf16 S[16][1024];          // scores / exp(scores)
  __shared__ float  red[16][8];
  __shared__ float  rstat[16];

  // ---- phase 0: stage Q tile (dw applied, bf16) with float4 loads ----
  for (int i = tid; i < 16 * 60; i += 128) {
    int row = i / 60, c4 = i - row * 60;
    int ch  = c4 * 4;
    const float4 x4 = *(const float4*)(xsin +
        (size_t)(b * NTOK + qt * 16 + row) * CTOT + h * 240 + ch);
    const float4 w4 = *(const float4*)(pwq + h * 240 + ch);
    const float4 b4 = *(const float4*)(pbq + h * 240 + ch);
    Qs[row][ch + 0] = (__bf16)(x4.x * w4.x + b4.x);
    Qs[row][ch + 1] = (__bf16)(x4.y * w4.y + b4.y);
    Qs[row][ch + 2] = (__bf16)(x4.z * w4.z + b4.z);
    Qs[row][ch + 3] = (__bf16)(x4.w * w4.w + b4.w);
  }
  for (int i = tid; i < 16 * 16; i += 128) Qs[i >> 4][240 + (i & 15)] = (__bf16)0.f;
  __syncthreads();

  // ---- phase 1: S = scale * Q @ K^T, waves split the 64 key tiles ----
  const int m  = lane & 15;
  const int lb = (lane >> 4) << 3;
  for (int j = wid; j < 64; j += 4) {
    int tok = j * 16 + m;                 // key token for B lane
    const float* row = xsin + (size_t)(b * NTOK + tok) * CTOT + h * 240;
    const float* wr  = pwk + h * 240;
    const float* br  = pbk + h * 240;
    __builtin_prefetch(row, 0, 1);
    v8f acc = {};
    for (int k0 = 0; k0 < 224; k0 += 32) {
      v16bf a  = fragA_lds(&Qs[m][0], k0, lane);
      v16bf bm = fragB_full(row + k0, wr + k0, br + k0, lb);
      acc = __builtin_amdgcn_wmma_f32_16x16x32_bf16(
          false, a, false, bm, (short)0, acc, false, false);
    }
    {
      v16bf a  = fragA_lds(&Qs[m][0], 224, lane);
      v16bf bm = fragB_tail(row + 224, wr + 224, br + 224, lb);
      acc = __builtin_amdgcn_wmma_f32_16x16x32_bf16(
          false, a, false, bm, (short)0, acc, false, false);
    }
    int col = j * 16 + m;
#pragma unroll
    for (int r = 0; r < 8; ++r) S[r + lb][col] = (__bf16)(acc[r] * scale);
  }
  __syncthreads();

  // ---- phase 2: row max / exp / row sum (1/sum folded into epilogue) ----
  {
    int row = tid & 15, part = tid >> 4;      // 8 parts x 128 cols
    float mx = -1e30f;
    for (int c = part * 128; c < part * 128 + 128; ++c)
      mx = fmaxf(mx, (float)S[row][c]);
    red[row][part] = mx;
    __syncthreads();
    if (tid < 16) {
      float m2 = red[tid][0];
#pragma unroll
      for (int p = 1; p < 8; ++p) m2 = fmaxf(m2, red[tid][p]);
      rstat[tid] = m2;
    }
    __syncthreads();
    float rmax = rstat[row];
    float psum = 0.f;
    for (int c = part * 128; c < part * 128 + 128; ++c) {
      float e = __expf((float)S[row][c] - rmax);
      S[row][c] = (__bf16)e;
      psum += e;
    }
    red[row][part] = psum;
    __syncthreads();
    if (tid < 16) {
      float s2 = 0.f;
#pragma unroll
      for (int p = 0; p < 8; ++p) s2 += red[tid][p];
      rstat[tid] = 1.f / s2;
    }
    __syncthreads();
  }

  // ---- phase 3: O = (P @ V) / rowsum ; each wave owns dv tiles ----
  int dvtiles = dv >> 4;                  // 1,2,4,8
  for (int dt = wid; dt < dvtiles; dt += 4) {
    int dvcol = dt * 16 + m;
    int chloc = h * dv + dvcol;           // group-local v channel
    const float* base = xsin + (size_t)(b * NTOK) * CTOT + coff + chloc;
    const float wvs = pwv[chloc], bvs = pbv[chloc];
    v8f acc = {};
    for (int k0 = 0; k0 < 1024; k0 += 32) {
      v16bf a = fragA_lds(&S[m][0], k0, lane);
      v16bf bm;
#pragma unroll
      for (int e = 0; e < 16; ++e) {
        int kk = k0 + kpat(e, lane);      // key token
        bm[e] = (__bf16)(base[(size_t)kk * CTOT] * wvs + bvs);
      }
      acc = __builtin_amdgcn_wmma_f32_16x16x32_bf16(
          false, a, false, bm, (short)0, acc, false, false);
    }
    int col = dt * 16 + m;
#pragma unroll
    for (int r = 0; r < 8; ++r) {
      int mm = r + lb;
      int n  = qt * 16 + mm;
      out[(size_t)(b * NTOK + n) * CTOT + coff + h * dv + col] =
          acc[r] * rstat[mm];
    }
  }
}

// ---------------------------------------------------------------------------
// out = xs2 + (0.5*ol + 0.5*og) * wproj + bproj   (spatial block residual)
// ---------------------------------------------------------------------------
__global__ void final_kernel(const float* __restrict__ xs2,
                             const float* __restrict__ og,
                             const float* __restrict__ ol,
                             const float* __restrict__ w,
                             const float* __restrict__ bia,
                             float* __restrict__ dout, int coff, int f) {
  int i = blockIdx.x * blockDim.x + threadIdx.x;
  int tot = BB * NTOK * f;
  if (i >= tot) return;
  int r = i / f, c = i - r * f;
  size_t k = (size_t)r * CTOT + coff + c;
  dout[i] = xs2[k] + (0.5f * ol[k] + 0.5f * og[k]) * w[c] + bia[c];
}

// ---------------------------------------------------------------------------
extern "C" void kernel_launch(void* const* d_in, const int* in_sizes, int n_in,
                              void* d_out, int out_size, void* d_ws,
                              size_t ws_size, hipStream_t stream) {
  (void)in_sizes; (void)n_in; (void)out_size; (void)ws_size;
  const int FEAT[4] = {64, 128, 256, 512};
  const int COFF[4] = {0, 64, 192, 448};
  const size_t NT = (size_t)BB * NTOK * CTOT;

  float* xcin = (float*)d_ws;      // LN output (concat); reused as xsin
  float* xs2  = xcin + NT;         // post-channel residual (concat layout)
  float* og   = xcin + 2 * NT;     // global-variant attention output
  float* ol   = xcin + 3 * NT;     // local-variant attention output

  // params: d_in[4 + g*32 + key*2 + {0:w,1:b}], keys in sorted-dict order
  enum { CKG = 0, CKL, CPROJ, CQG, CQL, CVG, CVL, LNC, LNS,
         SKG, SKL, SPROJ, SQG, SQL, SVG, SVL };
  auto P = [&](int g, int key, int wb) -> const float* {
    return (const float*)d_in[4 + g * 32 + key * 2 + wb];
  };

  const int rows = BB * NTOK;

  // ---- channel block: LN ----
  for (int g = 0; g < 4; ++g)
    ln_kernel<<<dim3((rows + 255) / 256), dim3(256), 0, stream>>>(
        (const float*)d_in[g], FEAT[g], P(g, LNC, 0), P(g, LNC, 1),
        xcin, COFF[g], FEAT[g]);

  // ---- channel attention (global + local variants) ----
  DWArgs ca;
  for (int g = 0; g < 4; ++g) {
    ca.wq[0][g] = P(g, CQG, 0); ca.bq[0][g] = P(g, CQG, 1);
    ca.wq[1][g] = P(g, CQL, 0); ca.bq[1][g] = P(g, CQL, 1);
    ca.wk[0][g] = P(g, CKG, 0); ca.bk[0][g] = P(g, CKG, 1);
    ca.wk[1][g] = P(g, CKL, 0); ca.bk[1][g] = P(g, CKL, 1);
    ca.wv[0][g] = P(g, CVG, 0); ca.bv[0][g] = P(g, CVG, 1);
    ca.wv[1][g] = P(g, CVL, 0); ca.bv[1][g] = P(g, CVL, 1);
  }
  chan_attn_kernel<<<dim3(15, 32), dim3(32), 0, stream>>>(ca, xcin, og, ol);

  // ---- channel projection + residual ----
  for (int g = 0; g < 4; ++g) {
    int tot = rows * FEAT[g];
    combine_chan_kernel<<<dim3((tot + 255) / 256), dim3(256), 0, stream>>>(
        (const float*)d_in[g], og, ol, P(g, CPROJ, 0), P(g, CPROJ, 1),
        xs2, COFF[g], FEAT[g]);
  }

  // ---- spatial block: LN (xcin buffer reused as xsin) ----
  for (int g = 0; g < 4; ++g)
    ln_kernel<<<dim3((rows + 255) / 256), dim3(256), 0, stream>>>(
        xs2 + COFF[g], CTOT, P(g, LNS, 0), P(g, LNS, 1),
        xcin, COFF[g], FEAT[g]);

  // ---- spatial attention (global + local variants) ----
  DWArgs sa;
  for (int g = 0; g < 4; ++g) {
    sa.wq[0][g] = P(g, SQG, 0); sa.bq[0][g] = P(g, SQG, 1);
    sa.wq[1][g] = P(g, SQL, 0); sa.bq[1][g] = P(g, SQL, 1);
    sa.wk[0][g] = P(g, SKG, 0); sa.bk[0][g] = P(g, SKG, 1);
    sa.wk[1][g] = P(g, SKL, 0); sa.bk[1][g] = P(g, SKL, 1);
    sa.wv[0][g] = P(g, SVG, 0); sa.bv[0][g] = P(g, SVG, 1);
    sa.wv[1][g] = P(g, SVL, 0); sa.bv[1][g] = P(g, SVL, 1);
  }
  spat_attn_kernel<<<dim3(64, 128), dim3(128), 0, stream>>>(sa, xcin, og, ol);

  // ---- spatial projection + residual -> d_out (groups concatenated) ----
  size_t ooff = 0;
  for (int g = 0; g < 4; ++g) {
    int tot = rows * FEAT[g];
    final_kernel<<<dim3((tot + 255) / 256), dim3(256), 0, stream>>>(
        xs2, og, ol, P(g, SPROJ, 0), P(g, SPROJ, 1),
        (float*)d_out + ooff, COFF[g], FEAT[g]);
    ooff += (size_t)tot;
  }
}
